// MPNNet_4355096838272
// MI455X (gfx1250) — matrix operand order repaired and checked by reading, not compiled, not run on hardware
//
#include <hip/hip_runtime.h>
#include <hip/hip_bf16.h>
#include <stdint.h>

#define N_NODES 100000
#define N_EDGES 1000000
#define NUM_G   1000
// D=64, ED=16, L=3, K_msg=80 (pad 96), K_upd=128, CAT=192

typedef __attribute__((ext_vector_type(16))) __bf16 v16bf;
typedef __attribute__((ext_vector_type(8)))  float  v8f;

union Frag16 { v16bf v; uint32_t u[8]; };
union AccF   { v8f   v; float    f[8]; };

__device__ __forceinline__ uint32_t pack_bf16(float a, float b) {
  uint32_t ua = __float_as_uint(a), ub = __float_as_uint(b);
  ua += 0x7FFFu + ((ua >> 16) & 1u);   // RNE to bf16
  ub += 0x7FFFu + ((ub >> 16) & 1u);
  return (ua >> 16) | (ub & 0xFFFF0000u);
}

// Async copy 16B from global to LDS (per-lane addresses), tracked by ASYNCcnt.
__device__ __forceinline__ void async_b128(uint32_t lds_addr, uint64_t gaddr) {
  asm volatile("global_load_async_to_lds_b128 %0, %1, off"
               :: "v"(lds_addr), "v"(gaddr) : "memory");
}
__device__ __forceinline__ uint32_t lds_u32(const void* p) {
  return (uint32_t)(uintptr_t)p;       // low 32 bits of flat addr = LDS byte offset
}

// ---------------------------------------------------------------------------
// Pack msgW (3 x 80x64, zero-pad K to 96) and updW (3 x 128x64) into WMMA
// B-fragment order: [chunk c][ntile t][vgpr v][lane]; per lane:
//   k = 32c + 16*(lane>>4) + 2v,  n = (lane&15) + 16t  (bf16 pair packed in u32)
// ---------------------------------------------------------------------------
__global__ void k_pack_weights(const float* __restrict__ msgW,
                               const float* __restrict__ updW,
                               uint32_t* __restrict__ wmsg,
                               uint32_t* __restrict__ wupd) {
  int i = blockIdx.x * 256 + threadIdx.x;
  if (i < 3 * 3072) {
    int l = i / 3072, r = i % 3072;
    int c = r >> 10, t = (r >> 8) & 3, v = (r >> 5) & 7, lane = r & 31;
    int k = 32 * c + 16 * (lane >> 4) + 2 * v;
    int n = (lane & 15) + 16 * t;
    float a = (k     < 80) ? msgW[(size_t)l * 5120 + (size_t)k       * 64 + n] : 0.f;
    float b = (k + 1 < 80) ? msgW[(size_t)l * 5120 + (size_t)(k + 1) * 64 + n] : 0.f;
    wmsg[i] = pack_bf16(a, b);
  }
  if (i < 3 * 4096) {
    int l = i / 4096, r = i % 4096;
    int c = r >> 10, t = (r >> 8) & 3, v = (r >> 5) & 7, lane = r & 31;
    int k = 32 * c + 16 * (lane >> 4) + 2 * v;
    int n = (lane & 15) + 16 * t;
    wupd[i] = pack_bf16(updW[(size_t)l * 8192 + (size_t)k       * 64 + n],
                        updW[(size_t)l * 8192 + (size_t)(k + 1) * 64 + n]);
  }
}

// edge_attr f32 -> packed bf16 (so edge staging can be a raw async LDS copy)
__global__ void k_pack_edge(const float* __restrict__ ea, uint32_t* __restrict__ eb) {
  size_t i = (size_t)blockIdx.x * 256 + threadIdx.x;     // E*8 packed dwords
  if (i >= (size_t)N_EDGES * 8) return;
  eb[i] = pack_bf16(ea[2 * i], ea[2 * i + 1]);
}

// x + vemb[l] row-bias, packed straight to bf16 (f32 copy not needed anywhere)
__global__ void k_add_vemb_pack(const float* __restrict__ xin, int stride,
                                const float* __restrict__ vrow,
                                uint32_t* __restrict__ xb) {
  int i = blockIdx.x * 256 + threadIdx.x;                // N*32 threads
  if (i >= N_NODES * 32) return;
  int n = i >> 5, j2 = i & 31;
  const float* xr = xin + (size_t)n * stride + 2 * j2;
  xb[(size_t)n * 32 + j2] =
      pack_bf16(xr[0] + vrow[2 * j2], xr[1] + vrow[2 * j2 + 1]);
}

// ---------------------------------------------------------------------------
// Edge message GEMM: m = relu([x[src] | edge_attr] @ msgW + msgb), atomically
// scattered into agg[dst]. One wave = 16 edges x 64 output cols; K=96 (3 chunks).
// All LDS staging (weights, gathered x rows, edge rows) via async-to-LDS.
// ---------------------------------------------------------------------------
__global__ void __launch_bounds__(256, 2) k_edge_msg(
    const uint32_t* __restrict__ xb, const uint32_t* __restrict__ eb,
    const int* __restrict__ ei, const uint32_t* __restrict__ wfrag_g,
    const float* __restrict__ msgb, float* __restrict__ agg) {
  __shared__ uint32_t wfrag[3072];        // 12 KB: this layer's B fragments
  __shared__ uint32_t astage[8][768];     // 24 KB: 8 waves x 16 rows x 48 dwords
  int tid = threadIdx.x;
  int w = tid >> 5, lane = tid & 31;

  { // async broadcast-load of B fragments into LDS (768 x 16B chunks)
    uint64_t g  = (uint64_t)(uintptr_t)wfrag_g;
    uint32_t l0 = lds_u32(&wfrag[0]);
    #pragma unroll
    for (int it = 0; it < 3; ++it) {
      uint32_t byt = (uint32_t)(it * 256 + tid) * 16u;
      async_b128(l0 + byt, g + byt);
    }
  }

  int tile = blockIdx.x * 8 + w;
  bool active = (tile < N_EDGES / 16);
  int base = tile * 16;
  if (active) {
    int row = lane >> 1, odd = lane & 1;   // 2 lanes per A row
    // gather x[src] row (128B of bf16) straight into LDS
    int s = ei[base + row];
    uint64_t gx = (uint64_t)(uintptr_t)(xb + (size_t)s * 32 + odd * 16);
    uint32_t lx = lds_u32(&astage[w][row * 48 + odd * 16]);
    #pragma unroll
    for (int it = 0; it < 4; ++it) async_b128(lx + it * 16u, gx + (uint64_t)it * 16u);
    // edge_attr row (32B of pre-packed bf16) -> dwords 32..39
    uint64_t ge = (uint64_t)(uintptr_t)(eb + (size_t)(base + row) * 8 + odd * 4);
    async_b128(lds_u32(&astage[w][row * 48 + 32 + odd * 4]), ge);
    // zero-pad K 80..95 (dwords 40..47)
    if (lane < 16) {
      uint4 z = {0u, 0u, 0u, 0u};
      *(uint4*)(&astage[w][lane * 48 + 40]) = z;
      *(uint4*)(&astage[w][lane * 48 + 44]) = z;
    }
  }
  asm volatile("s_wait_asynccnt 0" ::: "memory");
  __syncthreads();                         // wfrag visible block-wide
  if (!active) return;
  asm volatile("s_wait_dscnt 0" ::: "memory");

  AccF acc[4];
  #pragma unroll
  for (int t = 0; t < 4; ++t)
    #pragma unroll
    for (int p = 0; p < 8; ++p) acc[t].f[p] = 0.f;

  int m = lane & 15, khalf = lane >> 4;
  const uint32_t* arow = &astage[w][m * 48];
  #pragma unroll
  for (int c = 0; c < 3; ++c) {
    Frag16 a;                              // A: lane-half K-offset 16
    int d0 = 16 * c + 8 * khalf;
    #pragma unroll
    for (int q = 0; q < 4; ++q) { a.u[q] = arow[d0 + q]; a.u[4 + q] = arow[d0 + 4 + q]; }
    #pragma unroll
    for (int t = 0; t < 4; ++t) {
      Frag16 b;
      const uint32_t* wp = &wfrag[(c * 4 + t) * 256 + lane];
      #pragma unroll
      for (int v = 0; v < 8; ++v) b.u[v] = wp[v * 32];
      acc[t].v = __builtin_amdgcn_wmma_f32_16x16x32_bf16(
          false, a.v, false, b.v, (short)0, acc[t].v, false, false);
    }
  }

  // bias + relu + scatter-add into agg[dst] (L2-resident f32 atomics)
  int dstm[8];
  #pragma unroll
  for (int p = 0; p < 8; ++p) dstm[p] = ei[N_EDGES + base + khalf * 8 + p];
  #pragma unroll
  for (int t = 0; t < 4; ++t) {
    int n = (lane & 15) + 16 * t;
    float bias = msgb[n];
    #pragma unroll
    for (int p = 0; p < 8; ++p) {
      float val = acc[t].f[p] + bias;
      if (val > 0.f) atomicAdd(&agg[(size_t)dstm[p] * 64 + n], val);
    }
  }
}

// ---------------------------------------------------------------------------
// Node update: h = [x | agg] @ updW + updb; x_out = relu(LN(h)).
// One wave = 16 nodes; K = 128 -> 4 bf16 WMMA chunks per n-tile.
// x half async-copied from packed xb; agg half converted f32->bf16 in-kernel.
// ---------------------------------------------------------------------------
__global__ void __launch_bounds__(256, 2) k_node_update(
    const uint32_t* __restrict__ xb, const float* __restrict__ agg,
    const uint32_t* __restrict__ wfrag_g, const float* __restrict__ updb,
    const float* __restrict__ lng, const float* __restrict__ lnb,
    float* __restrict__ xcat, int colofs) {
  __shared__ uint32_t wfrag[4096];        // 16 KB
  __shared__ uint32_t astage[8][1024];    // 32 KB: 16 rows x 64 dwords (128 bf16)
  int tid = threadIdx.x;
  int w = tid >> 5, lane = tid & 31;

  { // async B fragments (1024 x 16B chunks)
    uint64_t g  = (uint64_t)(uintptr_t)wfrag_g;
    uint32_t l0 = lds_u32(&wfrag[0]);
    #pragma unroll
    for (int it = 0; it < 4; ++it) {
      uint32_t byt = (uint32_t)(it * 256 + tid) * 16u;
      async_b128(l0 + byt, g + byt);
    }
  }

  int tile = blockIdx.x * 8 + w;
  bool active = (tile < N_NODES / 16);
  int base = tile * 16;
  if (active) {
    int row = lane >> 1, odd = lane & 1;
    int n = base + row;
    // K 0..63: x row already bf16 -> async straight into LDS
    uint64_t gx = (uint64_t)(uintptr_t)(xb + (size_t)n * 32 + odd * 16);
    uint32_t lx = lds_u32(&astage[w][row * 64 + odd * 16]);
    #pragma unroll
    for (int it = 0; it < 4; ++it) async_b128(lx + it * 16u, gx + (uint64_t)it * 16u);
    // K 64..127: agg row f32 -> bf16 pack (dwords 32..63)
    const float2* s2 = (const float2*)(agg + (size_t)n * 64 + odd * 32);
    uint32_t* dp = &astage[w][row * 64 + 32 + odd * 16];
    #pragma unroll
    for (int q = 0; q < 16; ++q) { float2 p = s2[q]; dp[q] = pack_bf16(p.x, p.y); }
  }
  asm volatile("s_wait_asynccnt 0" ::: "memory");
  __syncthreads();
  if (!active) return;
  asm volatile("s_wait_dscnt 0" ::: "memory");

  AccF acc[4];
  #pragma unroll
  for (int t = 0; t < 4; ++t)
    #pragma unroll
    for (int p = 0; p < 8; ++p) acc[t].f[p] = 0.f;

  int r = lane & 15, khalf = lane >> 4;
  const uint32_t* arow = &astage[w][r * 64];
  #pragma unroll
  for (int c = 0; c < 4; ++c) {
    Frag16 a;
    int d0 = 16 * c + 8 * khalf;
    #pragma unroll
    for (int q = 0; q < 4; ++q) { a.u[q] = arow[d0 + q]; a.u[4 + q] = arow[d0 + 4 + q]; }
    #pragma unroll
    for (int t = 0; t < 4; ++t) {
      Frag16 b;
      const uint32_t* wp = &wfrag[(c * 4 + t) * 256 + lane];
      #pragma unroll
      for (int v = 0; v < 8; ++v) b.u[v] = wp[v * 32];
      acc[t].v = __builtin_amdgcn_wmma_f32_16x16x32_bf16(
          false, a.v, false, b.v, (short)0, acc[t].v, false, false);
    }
  }

  // h tile back to LDS (reuse astage as f32), then LayerNorm+ReLU per row
  asm volatile("s_wait_dscnt 0" ::: "memory");
  float* hbase = (float*)&astage[w][0];
  #pragma unroll
  for (int t = 0; t < 4; ++t) {
    int n = (lane & 15) + 16 * t;
    float bias = updb[n];
    #pragma unroll
    for (int p = 0; p < 8; ++p) {
      int mm = p + 8 * khalf;               // C layout: VGPR p, lane-half -> row
      hbase[mm * 64 + n] = acc[t].f[p] + bias;
    }
  }
  asm volatile("s_wait_dscnt 0" ::: "memory");
  if (lane < 16) {
    const float* hr = hbase + lane * 64;
    float s = 0.f, s2 = 0.f;
    #pragma unroll
    for (int j = 0; j < 64; ++j) { float h = hr[j]; s += h; s2 += h * h; }
    float mu = s * (1.f / 64.f);
    float var = s2 * (1.f / 64.f) - mu * mu;
    float rstd = rsqrtf(var + 1e-5f);
    float* outr = xcat + (size_t)(base + lane) * 192 + colofs;
    #pragma unroll
    for (int j = 0; j < 64; ++j) {
      float y = (hr[j] - mu) * rstd * lng[j] + lnb[j];
      outr[j] = fmaxf(y, 0.f);
    }
  }
}

// JumpingKnowledge 'cat' already materialized in xcat; global_add_pool:
__global__ void k_pool(const float* __restrict__ xcat, const int* __restrict__ batch,
                       float* __restrict__ g) {
  size_t i = (size_t)blockIdx.x * 256 + threadIdx.x;
  if (i >= (size_t)N_NODES * 192) return;
  int n = (int)(i / 192), c = (int)(i % 192);
  atomicAdd(&g[(size_t)batch[n] * 192 + c], xcat[i]);
}

__global__ void k_linear(const float* __restrict__ in, const float* __restrict__ W,
                         const float* __restrict__ bias, float* __restrict__ out,
                         int K, int Nout, int relu) {
  int i = blockIdx.x * 256 + threadIdx.x;
  if (i >= NUM_G * Nout) return;
  int gi = i / Nout, j = i % Nout;
  const float* ir = in + (size_t)gi * K;
  float s = bias[j];
  for (int k = 0; k < K; ++k) s += ir[k] * W[(size_t)k * Nout + j];
  if (relu) s = fmaxf(s, 0.f);
  out[i] = s;
}

__global__ void k_out(const float* __restrict__ emb, const float* __restrict__ outW,
                      const float* __restrict__ outb, float* __restrict__ out) {
  int gi = blockIdx.x * 256 + threadIdx.x;
  if (gi >= NUM_G) return;
  const float* er = emb + (size_t)gi * 128;
  float s = outb[0];
  for (int k = 0; k < 128; ++k) s += er[k] * outW[k];
  out[gi] = s;
}

extern "C" void kernel_launch(void* const* d_in, const int* in_sizes, int n_in,
                              void* d_out, int out_size, void* d_ws, size_t ws_size,
                              hipStream_t stream) {
  const float* x         = (const float*)d_in[0];
  const float* edge_attr = (const float*)d_in[1];
  const int*   ei        = (const int*)d_in[2];
  const int*   batch     = (const int*)d_in[3];
  const float* vemb      = (const float*)d_in[5];
  const float* msgW      = (const float*)d_in[6];
  const float* msgb      = (const float*)d_in[7];
  const float* updW      = (const float*)d_in[8];
  const float* updb      = (const float*)d_in[9];
  const float* lng       = (const float*)d_in[10];
  const float* lnb       = (const float*)d_in[11];
  // d_in[12..17] = virtual-node MLP params: dead code in the reference (skipped)
  const float* linW0 = (const float*)d_in[18];
  const float* linb0 = (const float*)d_in[19];
  const float* linW1 = (const float*)d_in[20];
  const float* linb1 = (const float*)d_in[21];
  const float* embW  = (const float*)d_in[22];
  const float* embb  = (const float*)d_in[23];
  const float* outWp = (const float*)d_in[24];
  const float* outbp = (const float*)d_in[25];
  float* out = (float*)d_out;

  float* ws = (float*)d_ws;
  size_t o = 0;
  float*    agg  = ws + o;               o += (size_t)N_NODES * 64;
  float*    xcat = ws + o;               o += (size_t)N_NODES * 192;
  uint32_t* xb   = (uint32_t*)(ws + o);  o += (size_t)N_NODES * 32;
  uint32_t* eb   = (uint32_t*)(ws + o);  o += (size_t)N_EDGES * 8;
  uint32_t* wmsg = (uint32_t*)(ws + o);  o += 3 * 3072;
  uint32_t* wupd = (uint32_t*)(ws + o);  o += 3 * 4096;
  float*    g    = ws + o;               o += (size_t)NUM_G * 192;
  float*    g1   = ws + o;               o += (size_t)NUM_G * 256;
  float*    g2   = ws + o;               o += (size_t)NUM_G * 128;
  float*    embv = ws + o;               o += (size_t)NUM_G * 128;

  hipLaunchKernelGGL(k_pack_weights, dim3(48), dim3(256), 0, stream, msgW, updW, wmsg, wupd);
  hipLaunchKernelGGL(k_pack_edge, dim3(31250), dim3(256), 0, stream, edge_attr, eb);
  hipMemsetAsync(g, 0, (size_t)NUM_G * 192 * sizeof(float), stream);

  for (int l = 0; l < 3; ++l) {
    const float* xin = (l == 0) ? x : (xcat + (size_t)(l - 1) * 64);
    int stride = (l == 0) ? 64 : 192;
    hipLaunchKernelGGL(k_add_vemb_pack, dim3(12500), dim3(256), 0, stream,
                       xin, stride, vemb + l * 64, xb);
    hipMemsetAsync(agg, 0, (size_t)N_NODES * 64 * sizeof(float), stream);
    hipLaunchKernelGGL(k_edge_msg, dim3(7813), dim3(256), 0, stream,
                       xb, eb, ei, wmsg + (size_t)l * 3072, msgb + l * 64, agg);
    hipLaunchKernelGGL(k_node_update, dim3(782), dim3(256), 0, stream,
                       xb, agg, wupd + (size_t)l * 4096, updb + l * 64,
                       lng + l * 64, lnb + l * 64, xcat, l * 64);
  }

  hipLaunchKernelGGL(k_pool, dim3(75000), dim3(256), 0, stream, xcat, batch, g);
  hipLaunchKernelGGL(k_linear, dim3((NUM_G * 256 + 255) / 256), dim3(256), 0, stream,
                     g, linW0, linb0, g1, 192, 256, 1);
  hipLaunchKernelGGL(k_linear, dim3((NUM_G * 128 + 255) / 256), dim3(256), 0, stream,
                     g1, linW1, linb1, g2, 256, 128, 1);
  hipLaunchKernelGGL(k_linear, dim3((NUM_G * 128 + 255) / 256), dim3(256), 0, stream,
                     g2, embW, embb, embv, 128, 128, 0);
  hipLaunchKernelGGL(k_out, dim3((NUM_G + 255) / 256), dim3(256), 0, stream,
                     embv, outWp, outbp, out);
}
